// UltimateFusionV4_13280038879563
// MI455X (gfx1250) — matrix operand order repaired and checked by reading, not compiled, not run on hardware
//
#include <hip/hip_runtime.h>
#include <hip/hip_bf16.h>
#include <math.h>

typedef __bf16 bf16;
typedef __attribute__((ext_vector_type(16))) __bf16 bf16x16;
typedef __attribute__((ext_vector_type(8)))  __bf16 bf16x8;
typedef __attribute__((ext_vector_type(8)))  float  f32x8;
typedef __attribute__((ext_vector_type(4)))  float  f32x4;

#define DIMD 1024
#define SEQ  512
#define MTOK 4096   // B*S
#define NBLK 16

// epilogue modes
#define EPI_ATT  0
#define EPI_GELU 1
#define EPI_TANH 2
#define EPI_FFN2 3
#define EPI_BIAS 4

union ABf { bf16x16 v16; bf16x8 v8[2]; };

__device__ __forceinline__ float sigmoidf_(float x) { return 1.f / (1.f + __expf(-x)); }

// ---------------------------------------------------------------------------
// Tiled bf16 WMMA GEMM: C[M,N] = A[M,K] (bf16, row-major) x W[K,N], where W is
// pre-transposed+cast to Bt[N][K] bf16 so B fragments are contiguous in K.
// BM=64, BN=128, BK=32; 256 threads = 8 waves, wave grid 2(m) x 4(n),
// each wave owns a 32x32 output patch = 2x2 v_wmma_f32_16x16x32_bf16 tiles.
// Double-buffered LDS, register prefetch, one barrier per K-step.
// ---------------------------------------------------------------------------
template <int EPI>
__global__ __launch_bounds__(256)
void gemm_wmma_bf16(const bf16* __restrict__ A, const bf16* __restrict__ Bt,
                    const float* __restrict__ bias_base,
                    int M, int N, int K,
                    const int* __restrict__ idx, int step,
                    const float* __restrict__ gate_base,
                    const float* __restrict__ torsion,
                    const float* __restrict__ xin,     // EPI_ATT residual source
                    const float* __restrict__ pmean,   // EPI_FFN2 cross-pathway mean
                    const float* __restrict__ h1in,    // EPI_FFN2 residual source
                    float* __restrict__ outF,
                    bf16* __restrict__ outH) {
  constexpr int BM = 64, BN = 128, BK = 32, LDP = BK + 8;
  __shared__ bf16 As[2][BM][LDP];
  __shared__ bf16 Bs[2][BN][LDP];

  const int tid  = threadIdx.x;
  const int lane = tid & 31;
  const int wave = tid >> 5;
  const int wm   = wave >> 2;   // 0..1
  const int wn   = wave & 3;    // 0..3
  const int bm0  = blockIdx.y * BM;
  const int bn0  = blockIdx.x * BN;

  const int blk = idx ? idx[step] : 0;
  const float* bias = bias_base + (size_t)blk * N;

  // staging: A tile 64x32 (16B/thread), B tile 128x32 (32B/thread)
  const int arow = tid >> 2, acol = (tid & 3) << 3;
  const int brow = tid >> 1, bcol = (tid & 1) << 4;
  const bf16* Arow = A  + (size_t)(bm0 + arow) * K;
  const bf16* Brow = Bt + (size_t)(bn0 + brow) * K;

  f32x8 acc[2][2];
#pragma unroll
  for (int i = 0; i < 2; i++)
#pragma unroll
    for (int j = 0; j < 2; j++)
#pragma unroll
      for (int e = 0; e < 8; e++) acc[i][j][e] = 0.f;

  {  // stage kt=0
    bf16x8 a0 = *(const bf16x8*)(Arow + acol);
    bf16x8 b0 = *(const bf16x8*)(Brow + bcol);
    bf16x8 b1 = *(const bf16x8*)(Brow + bcol + 8);
    *(bf16x8*)&As[0][arow][acol]     = a0;
    *(bf16x8*)&Bs[0][brow][bcol]     = b0;
    *(bf16x8*)&Bs[0][brow][bcol + 8] = b1;
  }
  __syncthreads();

  const int KT = K / BK;
  bf16x8 ra, rb0, rb1;
  if (KT > 1) {
    ra  = *(const bf16x8*)(Arow + BK + acol);
    rb0 = *(const bf16x8*)(Brow + BK + bcol);
    rb1 = *(const bf16x8*)(Brow + BK + bcol + 8);
  }

  const int l16 = lane & 15, half = lane >> 4;
  for (int kt = 0; kt < KT; ++kt) {
    const int cur = kt & 1;
    // A frag: lanes 0-15 -> K 0..7 & 16..23, lanes 16-31 -> K 8..15 & 24..31
    ABf af[2], bf_[2];
#pragma unroll
    for (int i = 0; i < 2; i++) {
      const int r = wm * 32 + i * 16 + l16;
      const int k0 = half * 8;
      af[i].v8[0] = *(const bf16x8*)&As[cur][r][k0];
      af[i].v8[1] = *(const bf16x8*)&As[cur][r][k0 + 16];
    }
    // B frag: lanes 0-15 -> K 0..7 & 16..23, lanes 16-31 -> K 8..15 & 24..31
#pragma unroll
    for (int j = 0; j < 2; j++) {
      const int c = wn * 32 + j * 16 + l16;
      const int k0 = half * 8;
      bf_[j].v8[0] = *(const bf16x8*)&Bs[cur][c][k0];
      bf_[j].v8[1] = *(const bf16x8*)&Bs[cur][c][k0 + 16];
    }
#pragma unroll
    for (int i = 0; i < 2; i++)
#pragma unroll
      for (int j = 0; j < 2; j++)
        acc[i][j] = __builtin_amdgcn_wmma_f32_16x16x32_bf16(
            false, af[i].v16, false, bf_[j].v16, (short)0, acc[i][j], false, false);

    if (kt + 1 < KT) {
      const int nxt = cur ^ 1;
      *(bf16x8*)&As[nxt][arow][acol]     = ra;
      *(bf16x8*)&Bs[nxt][brow][bcol]     = rb0;
      *(bf16x8*)&Bs[nxt][brow][bcol + 8] = rb1;
      if (kt + 2 < KT) {
        const size_t ko = (size_t)(kt + 2) * BK;
        ra  = *(const bf16x8*)(Arow + ko + acol);
        rb0 = *(const bf16x8*)(Brow + ko + bcol);
        rb1 = *(const bf16x8*)(Brow + ko + bcol + 8);
      }
    }
    __syncthreads();
  }

  // epilogue; C layout: VGPR v -> row = base + (lane>>4)*8 + v, col = base + (lane&15)
  const float* gate = nullptr;
  if constexpr (EPI == EPI_ATT || EPI == EPI_FFN2) gate = gate_base + (size_t)blk * DIMD;

#pragma unroll
  for (int i = 0; i < 2; i++) {
#pragma unroll
    for (int j = 0; j < 2; j++) {
      const int rbase = bm0 + wm * 32 + i * 16 + half * 8;
      const int c = bn0 + wn * 32 + j * 16 + l16;
      const float bv = bias[c];
#pragma unroll
      for (int v = 0; v < 8; ++v) {
        const int r = rbase + v;
        const float val = acc[i][j][v] + bv;
        const size_t o = (size_t)r * N + c;
        if constexpr (EPI == EPI_ATT) {
          const int bb = r >> 9;  // S = 512
          const float g = sigmoidf_(gate[c]);
          const float t = val * (1.f + g * torsion[bb * DIMD + c]);
          outF[o] = xin[o] + 0.5f * t;
        } else if constexpr (EPI == EPI_GELU) {
          outH[o] = (bf16)(0.5f * val * (1.f + erff(val * 0.70710678118f)));
        } else if constexpr (EPI == EPI_TANH) {
          outH[o] = (bf16)tanhf(val);
        } else if constexpr (EPI == EPI_FFN2) {
          const int bb = r >> 9;
          const float g = sigmoidf_(gate[c]);
          const float h2 = val + pmean[bb * DIMD + c] * 0.3f * g;
          outF[o] = h1in[o] + 0.5f * h2;
        } else {
          outF[o] = val;
        }
      }
    }
  }
}

// LayerNorm * w + b -> bf16, one 256-thread block per token row (D=1024)
__global__ __launch_bounds__(256)
void ln_affine_bf16(const float* __restrict__ x,
                    const float* __restrict__ wbase, const float* __restrict__ bbase,
                    const int* __restrict__ idx, int step, bf16* __restrict__ out) {
  __shared__ float red[256];
  const int row = blockIdx.x, t = threadIdx.x;
  const int blk = idx[step];
  const float* w  = wbase + (size_t)blk * DIMD;
  const float* bb = bbase + (size_t)blk * DIMD;
  const float* xr = x + (size_t)row * DIMD;

  f32x4 v = *(const f32x4*)(xr + t * 4);
  red[t] = v[0] + v[1] + v[2] + v[3];
  __syncthreads();
  for (int o = 128; o > 0; o >>= 1) { if (t < o) red[t] += red[t + o]; __syncthreads(); }
  const float mean = red[0] * (1.f / DIMD);
  __syncthreads();
  const float d0 = v[0] - mean, d1 = v[1] - mean, d2 = v[2] - mean, d3 = v[3] - mean;
  red[t] = d0 * d0 + d1 * d1 + d2 * d2 + d3 * d3;
  __syncthreads();
  for (int o = 128; o > 0; o >>= 1) { if (t < o) red[t] += red[t + o]; __syncthreads(); }
  const float inv = rsqrtf(red[0] * (1.f / DIMD) + 1e-5f);
  const int c0 = t * 4;
  bf16* orow = out + (size_t)row * DIMD + c0;
  orow[0] = (bf16)(d0 * inv * w[c0 + 0] + bb[c0 + 0]);
  orow[1] = (bf16)(d1 * inv * w[c0 + 1] + bb[c0 + 1]);
  orow[2] = (bf16)(d2 * inv * w[c0 + 2] + bb[c0 + 2]);
  orow[3] = (bf16)(d3 * inv * w[c0 + 3] + bb[c0 + 3]);
}

// f32 W[K][N] -> bf16 Wt[N][K] (per-block slice via idx), 32x32 LDS tiles
__global__ __launch_bounds__(256)
void transpose_cast_bf16(const float* __restrict__ W, bf16* __restrict__ out,
                         int K, int N, const int* __restrict__ idx, int step) {
  __shared__ float tile[32][33];
  const float* Wb = W + (idx ? (size_t)idx[step] * (size_t)K * N : 0);
  const int kb = blockIdx.x * 32, nb = blockIdx.y * 32;
  const int tx = threadIdx.x & 31, ty = threadIdx.x >> 5;
#pragma unroll
  for (int p = 0; p < 4; p++)
    tile[ty + p * 8][tx] = Wb[(size_t)(kb + ty + p * 8) * N + nb + tx];
  __syncthreads();
#pragma unroll
  for (int p = 0; p < 4; p++)
    out[(size_t)(nb + ty + p * 8) * K + kb + tx] = (bf16)tile[tx][ty + p * 8];
}

__global__ void seq_mean_f32(const float* __restrict__ x, float* __restrict__ out) {
  const int i = blockIdx.x * 256 + threadIdx.x;  // over B*D = 8192
  if (i >= 8 * DIMD) return;
  const int b = i >> 10, d = i & (DIMD - 1);
  const float* p = x + (size_t)b * SEQ * DIMD + d;
  float s = 0.f;
  for (int ss = 0; ss < SEQ; ++ss) s += p[(size_t)ss * DIMD];
  out[i] = s * (1.f / SEQ);
}

// sigmoid(cmean0 @ selW + selB)*0.6 + prior*0.4 on batch elem 0; top-4 per path
__global__ void routing_topk(const float* __restrict__ cmean, const float* __restrict__ bmean,
                             const float* __restrict__ scw, const float* __restrict__ scb,
                             const float* __restrict__ pc,
                             const float* __restrict__ sbw, const float* __restrict__ sbb,
                             const float* __restrict__ pb, int* __restrict__ idx_out) {
  __shared__ float adj[32];
  const int t = threadIdx.x;
  if (t < 32) {
    const int j = t & 15;
    const float* m  = (t < 16) ? cmean : bmean;
    const float* w  = (t < 16) ? scw : sbw;
    const float* sb = (t < 16) ? scb : sbb;
    const float* pr = (t < 16) ? pc : pb;
    float a = sb[j];
    for (int d = 0; d < DIMD; ++d) a += m[d] * w[d * NBLK + j];
    adj[t] = sigmoidf_(a) * 0.6f + pr[j] * 0.4f;
  }
  __syncthreads();
  if (t == 0 || t == 16) {
    const float* a = &adj[t];
    const int base = (t == 0) ? 0 : 4;
    bool used[16] = {false};
    for (int k = 0; k < 4; k++) {
      int best = 0; float bv = -1e30f;
      for (int j = 0; j < NBLK; j++)
        if (!used[j] && a[j] > bv) { bv = a[j]; best = j; }
      used[best] = true;
      idx_out[base + k] = best;
    }
  }
}

__global__ void copy_f32(const float* __restrict__ in, float* __restrict__ out, int n4) {
  const int i = blockIdx.x * 256 + threadIdx.x;
  if (i < n4) ((f32x4*)out)[i] = ((const f32x4*)in)[i];
}

__global__ void concat_cast_bf16(const float* __restrict__ hc, const float* __restrict__ hb,
                                 bf16* __restrict__ out) {
  const size_t i = (size_t)blockIdx.x * 256 + threadIdx.x;  // over M*2048
  const size_t row = i >> 11;
  const int c = (int)(i & 2047);
  const float v = (c < DIMD) ? hc[row * DIMD + c] : hb[row * DIMD + (c - DIMD)];
  out[i] = (bf16)v;
}

extern "C" void kernel_launch(void* const* d_in, const int* in_sizes, int n_in,
                              void* d_out, int out_size, void* d_ws, size_t ws_size,
                              hipStream_t stream) {
  const float* cortical  = (const float*)d_in[0];
  const float* brainstem = (const float*)d_in[1];
  const float* torsion   = (const float*)d_in[2];
  const float* prior_c   = (const float*)d_in[3];
  const float* prior_b   = (const float*)d_in[4];
  const float* c_n1w = (const float*)d_in[5];
  const float* c_n1b = (const float*)d_in[6];
  const float* c_aw  = (const float*)d_in[7];
  const float* c_ab  = (const float*)d_in[8];
  const float* c_n2w = (const float*)d_in[9];
  const float* c_n2b = (const float*)d_in[10];
  const float* c_f1w = (const float*)d_in[11];
  const float* c_f1b = (const float*)d_in[12];
  const float* c_f2w = (const float*)d_in[13];
  const float* c_f2b = (const float*)d_in[14];
  const float* c_gate = (const float*)d_in[15];
  const float* b_n1w = (const float*)d_in[16];
  const float* b_n1b = (const float*)d_in[17];
  const float* b_aw  = (const float*)d_in[18];
  const float* b_ab  = (const float*)d_in[19];
  const float* b_n2w = (const float*)d_in[20];
  const float* b_n2b = (const float*)d_in[21];
  const float* b_f1w = (const float*)d_in[22];
  const float* b_f1b = (const float*)d_in[23];
  const float* b_f2w = (const float*)d_in[24];
  const float* b_f2b = (const float*)d_in[25];
  const float* b_gate = (const float*)d_in[26];
  const float* sel_c_w = (const float*)d_in[27];
  const float* sel_c_b = (const float*)d_in[28];
  const float* sel_b_w = (const float*)d_in[29];
  const float* sel_b_b = (const float*)d_in[30];
  const float* cross_w = (const float*)d_in[31];
  const float* cross_b = (const float*)d_in[32];

  const size_t MD = (size_t)MTOK * DIMD;
  float* h_c   = (float*)d_out;        // output 0: cortical_h  (also running state)
  float* h_b   = h_c + MD;             // output 1: brainstem_h (also running state)
  float* fused = h_b + MD;             // output 2: fused

  char* wsb = (char*)d_ws;
  float* cmean = (float*)(wsb);
  float* bmean = (float*)(wsb + (64ull << 10));
  int*   idx   = (int*)(wsb + (128ull << 10));
  float* h1    = (float*)(wsb + (1ull  << 20));   // 16 MB
  bf16*  normb = (bf16*)(wsb + (17ull << 20));    //  8 MB
  bf16*  actb  = (bf16*)(wsb + (26ull << 20));    // 16 MB
  bf16*  btw   = (bf16*)(wsb + (43ull << 20));    //  4 MB (transposed bf16 weights)
  bf16*  catb  = (bf16*)(wsb + (48ull << 20));    // 16 MB

  const dim3 blk256(256);
  copy_f32<<<(unsigned)(MD / 4 / 256), blk256, 0, stream>>>(cortical, h_c, (int)(MD / 4));
  copy_f32<<<(unsigned)(MD / 4 / 256), blk256, 0, stream>>>(brainstem, h_b, (int)(MD / 4));
  seq_mean_f32<<<(8 * DIMD) / 256, blk256, 0, stream>>>(cortical, cmean);
  seq_mean_f32<<<(8 * DIMD) / 256, blk256, 0, stream>>>(brainstem, bmean);
  routing_topk<<<1, 32, 0, stream>>>(cmean, bmean, sel_c_w, sel_c_b, prior_c,
                                     sel_b_w, sel_b_b, prior_b, idx);

  const dim3 gN1(DIMD / 128, MTOK / 64);   // N=1024
  const dim3 gN2(2048 / 128, MTOK / 64);   // N=2048
  const dim3 gT11(DIMD / 32, DIMD / 32);   // K=1024,N=1024
  const dim3 gT12(DIMD / 32, 2048 / 32);   // K=1024,N=2048
  const dim3 gT21(2048 / 32, DIMD / 32);   // K=2048,N=1024

  for (int step = 0; step < 4; ++step) {   // cortical pathway: idx[0..3]
    const int st = step;
    ln_affine_bf16<<<MTOK, blk256, 0, stream>>>(h_c, c_n1w, c_n1b, idx, st, normb);
    transpose_cast_bf16<<<gT11, blk256, 0, stream>>>(c_aw, btw, DIMD, DIMD, idx, st);
    gemm_wmma_bf16<EPI_ATT><<<gN1, blk256, 0, stream>>>(normb, btw, c_ab, MTOK, DIMD, DIMD,
        idx, st, c_gate, torsion, h_c, nullptr, nullptr, h1, nullptr);
    ln_affine_bf16<<<MTOK, blk256, 0, stream>>>(h1, c_n2w, c_n2b, idx, st, normb);
    transpose_cast_bf16<<<gT12, blk256, 0, stream>>>(c_f1w, btw, DIMD, 2048, idx, st);
    gemm_wmma_bf16<EPI_GELU><<<gN2, blk256, 0, stream>>>(normb, btw, c_f1b, MTOK, 2048, DIMD,
        idx, st, nullptr, nullptr, nullptr, nullptr, nullptr, nullptr, actb);
    transpose_cast_bf16<<<gT21, blk256, 0, stream>>>(c_f2w, btw, 2048, DIMD, idx, st);
    gemm_wmma_bf16<EPI_FFN2><<<gN1, blk256, 0, stream>>>(actb, btw, c_f2b, MTOK, DIMD, 2048,
        idx, st, c_gate, nullptr, nullptr, bmean, h1, h_c, nullptr);
  }
  for (int step = 0; step < 4; ++step) {   // brainstem pathway: idx[4..7]
    const int st = 4 + step;
    ln_affine_bf16<<<MTOK, blk256, 0, stream>>>(h_b, b_n1w, b_n1b, idx, st, normb);
    transpose_cast_bf16<<<gT11, blk256, 0, stream>>>(b_aw, btw, DIMD, DIMD, idx, st);
    gemm_wmma_bf16<EPI_ATT><<<gN1, blk256, 0, stream>>>(normb, btw, b_ab, MTOK, DIMD, DIMD,
        idx, st, b_gate, torsion, h_b, nullptr, nullptr, h1, nullptr);
    ln_affine_bf16<<<MTOK, blk256, 0, stream>>>(h1, b_n2w, b_n2b, idx, st, normb);
    transpose_cast_bf16<<<gT11, blk256, 0, stream>>>(b_f1w, btw, DIMD, DIMD, idx, st);
    gemm_wmma_bf16<EPI_TANH><<<gN1, blk256, 0, stream>>>(normb, btw, b_f1b, MTOK, DIMD, DIMD,
        idx, st, nullptr, nullptr, nullptr, nullptr, nullptr, nullptr, actb);
    transpose_cast_bf16<<<gT11, blk256, 0, stream>>>(b_f2w, btw, DIMD, DIMD, idx, st);
    gemm_wmma_bf16<EPI_FFN2><<<gN1, blk256, 0, stream>>>(actb, btw, b_f2b, MTOK, DIMD, DIMD,
        idx, st, b_gate, nullptr, nullptr, cmean, h1, h_b, nullptr);
  }

  concat_cast_bf16<<<(unsigned)(MD * 2 / 256), blk256, 0, stream>>>(h_c, h_b, catb);
  transpose_cast_bf16<<<gT21, blk256, 0, stream>>>(cross_w, btw, 2048, DIMD, nullptr, 0);
  gemm_wmma_bf16<EPI_BIAS><<<gN1, blk256, 0, stream>>>(catb, btw, cross_b, MTOK, DIMD, 2048,
      nullptr, 0, nullptr, nullptr, nullptr, nullptr, nullptr, fused, nullptr);
}
